// Mamba3Layer_48962627174671
// MI455X (gfx1250) — matrix operand order repaired and checked by reading, not compiled, not run on hardware
//
#include <hip/hip_runtime.h>

// ---------------------------------------------------------------------------
// Mamba3-style chunked SSM layer for MI455X (gfx1250), wave32 + WMMA bf16.
// All heavy matmuls use v_wmma_f32_16x16x32_bf16 (bf16 in, f32 accumulate).
// ---------------------------------------------------------------------------

typedef __attribute__((ext_vector_type(16))) __bf16 v16bf;
typedef __attribute__((ext_vector_type(8)))  float  v8f;
typedef unsigned short u16;
typedef unsigned int   u32;

#define D_MODEL  1024
#define D_INNER  2048
#define N_HEADS  32
#define HEAD_DIM 64
#define D_STATE  64
#define CHUNK_L  128
#define D_PROJ   6432
#define BB       2
#define TT       2048
#define BT       (BB*TT)
#define NCH      (TT/CHUNK_L)
#define SCALE_F  0.08838834764831845f   /* (D_STATE*R)^-0.5 = 1/sqrt(128) */
#define EPS_F    1e-6f

// ---------------- helpers ----------------
__device__ __forceinline__ u16 f2bf(float f){
  union { float f; u32 u; } x; x.f = f;
  u32 r = x.u + 0x7fffu + ((x.u >> 16) & 1u);
  return (u16)(r >> 16);
}
__device__ __forceinline__ float bf2f(u16 v){
  union { u32 u; float f; } x; x.u = ((u32)v) << 16; return x.f;
}
__device__ __forceinline__ float sigmoidf_(float x){ return 1.f/(1.f+__expf(-x)); }
__device__ __forceinline__ float siluf_(float x){ return x*sigmoidf_(x); }

union Frag { v16bf v; u32 u[8]; };

// A-matrix 16x32 bf16 fragment (ISA 7.12.2): lanes 0-15 row M=lane,
// vgpr v holds K = {0..7,16..23}+ (lanes 16-31: +8 / +24).
__device__ __forceinline__ v16bf ldsFragA(const u16* p0, int stride, int row, int koff, int hl){
  Frag f; const u16* p = p0 + row*stride + koff + (hl ? 8 : 0);
#pragma unroll
  for (int i=0;i<8;i++){
    int k2 = ((i&4)?16:0) + ((i&3)<<1);
    f.u[i] = *(const u32*)(p + k2);
  }
  return f.v;
}
// B-matrix 32x16 bf16 fragment: lane = column, same K order as A:
// K = {0..7,16..23} (lanes 16-31: +8 / +24). p0 points at B^T storage: [col][k], k contiguous.
__device__ __forceinline__ v16bf ldsFragB(const u16* p0, int stride, int col, int koff, int hl){
  Frag f; const u16* p = p0 + col*stride + koff + (hl ? 8 : 0);
#pragma unroll
  for (int i=0;i<8;i++) f.u[i] = *(const u32*)(p + (((i&4)?16:0) + ((i&3)<<1)));
  return f.v;
}
__device__ __forceinline__ v8f zero8(){ v8f z; for(int i=0;i<8;i++) z[i]=0.f; return z; }
__device__ __forceinline__ v8f wmma_bf16(v16bf a, v16bf b, v8f c){
  return __builtin_amdgcn_wmma_f32_16x16x32_bf16(false, a, false, b, (short)0, c, false, false);
}

// ---------------- fp32 -> bf16 convert ----------------
__global__ __launch_bounds__(256) void cvt_bf16_kernel(const float* __restrict__ s,
                                                       u16* __restrict__ d, int n){
  int i = blockIdx.x*256 + threadIdx.x;
  if (i < n) d[i] = f2bf(s[i]);
}

// ---------------- 256x128 block WMMA GEMM (bf16 x bf16 -> f32) -------------
// A: [M][K] row-major bf16, B: [K][N] row-major bf16, C: [M][N] f32.
// M%256==0, K%32==0; N guarded. 8 waves, each owns a 64x64 tile:
// 16 WMMAs per K-step against 8 fragment loads (2x matrix density vs 32x64).
__global__ __launch_bounds__(256) void gemm_bf16_kernel(const u16* __restrict__ A,
    const u16* __restrict__ Bm, float* __restrict__ C, int M, int N, int K)
{
  __shared__ __align__(16) u16 As[256*40];   // [row][k], padded
  __shared__ __align__(16) u16 Bs[128*40];   // B^T: [n][k]
  const int tid = threadIdx.x;
  const int wv = tid>>5, lane = tid&31, hl = lane>>4, lr = lane&15;
  const int bm = blockIdx.y*256, bn = blockIdx.x*128;
  const int wm = (wv>>1)*64, wn = (wv&1)*64;   // 8 waves: 4(M) x 2(N)
  v8f acc[4][4];
  for (int i=0;i<4;i++) for (int j=0;j<4;j++) acc[i][j]=zero8();

  for (int k0=0; k0<K; k0+=32){
    if (k0+32 < K) __builtin_prefetch(&A[(size_t)bm*K + k0 + 32], 0, 0);
    // A tile 256x32, 16B vector loads (K, k0, c8 all keep 16B alignment)
    for (int i=tid;i<1024;i+=256){
      int r = i>>2, c8 = (i&3)<<3;
      *(uint4*)&As[r*40 + c8] = *(const uint4*)&A[(size_t)(bm+r)*K + k0 + c8];
    }
    // B tile 32x128: vectorized row read, scalar transpose store to Bs[n][k]
    for (int i=tid;i<512;i+=256){
      int kk = i>>4, n8 = (i&15)<<3;
      int gn = bn + n8;
      if (gn + 8 <= N){
        uint4 v = *(const uint4*)&Bm[(size_t)(k0+kk)*N + gn];
        const u16* e = (const u16*)&v;
#pragma unroll
        for (int j=0;j<8;j++) Bs[(n8+j)*40 + kk] = e[j];
      } else {
#pragma unroll
        for (int j=0;j<8;j++)
          Bs[(n8+j)*40 + kk] = (gn+j < N) ? Bm[(size_t)(k0+kk)*N + gn + j] : (u16)0;
      }
    }
    __syncthreads();
    v16bf af[4], bfv[4];
    for (int i=0;i<4;i++) af[i]  = ldsFragA(As,40, wm+i*16+lr, 0, hl);
    for (int j=0;j<4;j++) bfv[j] = ldsFragB(Bs,40, wn+j*16+lr, 0, hl);
    for (int i=0;i<4;i++)
      for (int j=0;j<4;j++)
        acc[i][j] = wmma_bf16(af[i], bfv[j], acc[i][j]);
    __syncthreads();
  }
  for (int i=0;i<4;i++)
    for (int j=0;j<4;j++){
#pragma unroll
      for (int jj=0;jj<8;jj++){
        int row = bm + wm + i*16 + jj + hl*8;   // C/D layout: M = jj + 8*half
        int col = bn + wn + j*16 + lr;
        if (col < N) C[(size_t)row*N + col] = acc[i][j][jj];
      }
    }
}

// ---------------- prep1: dt/lam/logdA, dtmean, RMS norms for B and C -------
__global__ __launch_bounds__(64) void prep1_kernel(const float* __restrict__ proj,
    const float* __restrict__ A_log, const float* __restrict__ dt_bias,
    const float* __restrict__ Bnb, const float* __restrict__ Cnb,
    float* __restrict__ dt_ws, float* __restrict__ lam_ws, float* __restrict__ ldA_ws,
    float* __restrict__ dtmean_ws, float* __restrict__ Bn_ws, float* __restrict__ Cn_ws)
{
  int bt = blockIdx.x, tid = threadIdx.x;
  const float* row = proj + (size_t)bt*D_PROJ;
  __shared__ float red[64];
  __shared__ float snorm;
  if (tid < 32){
    float draw = row[6336+tid] + dt_bias[tid];
    float sp = (draw > 20.f) ? draw : log1pf(__expf(draw));
    float dt = fminf(fmaxf(sp, 1e-4f), 0.5f);
    dt_ws[bt*32+tid]  = dt;
    lam_ws[bt*32+tid] = sigmoidf_(row[6368+tid]);
    ldA_ws[bt*32+tid] = dt * (-__expf(A_log[tid]));
    red[tid] = dt;
  } else red[tid] = 0.f;
  __syncthreads();
  if (tid==0){ float s=0.f; for(int i=0;i<32;i++) s+=red[i]; dtmean_ws[bt]=s*(1.f/32.f); }
  __syncthreads();
  float cv = row[6272+tid];
  red[tid] = cv*cv;
  __syncthreads();
  if (tid==0){ float s=0.f; for(int i=0;i<64;i++) s+=red[i]; snorm = rsqrtf(s*(1.f/64.f)+EPS_F); }
  __syncthreads();
  Cn_ws[bt*64+tid] = cv*snorm*Cnb[tid];
  for (int r=0;r<2;r++){
    __syncthreads();
    float bv = row[6144 + r*64 + tid];
    red[tid] = bv*bv;
    __syncthreads();
    if (tid==0){ float s=0.f; for(int i=0;i<64;i++) s+=red[i]; snorm = rsqrtf(s*(1.f/64.f)+EPS_F); }
    __syncthreads();
    Bn_ws[bt*128 + r*64 + tid] = bv*snorm*Bnb[tid];
  }
}

// ---------------- RoPE angle cumsum over full T (serial per batch) ---------
__global__ void ang_kernel(const float* __restrict__ proj, const float* __restrict__ dtmean,
                           float* __restrict__ cosw, float* __restrict__ sinw){
  int b = blockIdx.x, j = threadIdx.x;   // 32 threads
  float acc = 0.f;
  for (int t=0;t<TT;t++){
    int td = b*TT + t;
    acc += dtmean[td]*proj[(size_t)td*D_PROJ + 6400 + j];
    float ang = -acc;
    cosw[td*32+j] = __cosf(ang);
    sinw[td*32+j] = __sinf(ang);
  }
}

// ---------------- prep2: RoPE(B,C) -> bf16, xg/xb -> bf16 ------------------
__global__ __launch_bounds__(128) void prep2_kernel(const float* __restrict__ proj,
    const float* __restrict__ Bn, const float* __restrict__ Cn,
    const float* __restrict__ cosw, const float* __restrict__ sinw,
    const float* __restrict__ dt_ws, const float* __restrict__ lam_ws,
    const float* __restrict__ ldA_ws,
    u16* __restrict__ Crb, u16* __restrict__ Brb,
    u16* __restrict__ xg, u16* __restrict__ xb)
{
  int bt = blockIdx.x, tid = threadIdx.x;
  __shared__ float cs[32], sn[32], dts[32], lams[32], alps[32];
  if (tid < 32){
    cs[tid]=cosw[bt*32+tid]; sn[tid]=sinw[bt*32+tid];
    dts[tid]=dt_ws[bt*32+tid]; lams[tid]=lam_ws[bt*32+tid];
    alps[tid]=__expf(ldA_ws[bt*32+tid]);
  }
  __syncthreads();
  if (tid < 64){
    float v;
    if (tid < 32) v = Cn[bt*64+tid]*cs[tid] - Cn[bt*64+tid+32]*sn[tid];
    else { int j=tid-32; v = Cn[bt*64+j]*sn[j] + Cn[bt*64+tid]*cs[j]; }
    Crb[bt*64+tid] = f2bf(v);
  }
  {
    int r = tid>>6, n = tid&63;
    const float* bp = Bn + (size_t)bt*128 + r*64;
    float v;
    if (n < 32) v = bp[n]*cs[n] - bp[n+32]*sn[n];
    else { int j=n-32; v = bp[j]*sn[j] + bp[n]*cs[j]; }
    Brb[(size_t)bt*128 + r*64 + n] = f2bf(v);
  }
  int tb = bt & (TT-1);   // t within batch
  for (int it=0; it<32; ++it){
    int idx = it*128 + tid;          // (h,r,p) flattened: h=idx>>7
    int h = idx>>7;
    float xr = proj[(size_t)bt*D_PROJ + D_INNER + idx];
    xg[(size_t)bt*4096 + idx] = f2bf(lams[h]*dts[h]*siluf_(xr));
    float xbv = 0.f;
    if (tb > 0){
      float xp = proj[(size_t)(bt-1)*D_PROJ + D_INNER + idx];
      xbv = (1.f-lams[h])*dts[h]*alps[h]*siluf_(xp);
    }
    xb[(size_t)bt*4096 + idx] = f2bf(xbv);
  }
}

// ---------------- within-chunk cumsum of log(dA) ---------------------------
__global__ void chunk_cum_kernel(const float* __restrict__ ldA, float* __restrict__ cum){
  int t0 = blockIdx.x*CHUNK_L, h = threadIdx.x;   // 32 threads
  float acc = 0.f;
  for (int l=0;l<CHUNK_L;l++){
    acc += ldA[(size_t)(t0+l)*32 + h];
    cum[(size_t)(t0+l)*32 + h] = acc;
  }
}

// ---------------- intra-chunk attention (per b,chunk,head) -----------------
// scores = C (128x64) x Bv^T, decay-mask+SCALE weighting, then W x xv.
__global__ __launch_bounds__(256) void intra_kernel(
    const u16* __restrict__ Crb, const u16* __restrict__ Brb,
    const u16* __restrict__ xg, const u16* __restrict__ xb,
    const float* __restrict__ cum, float* __restrict__ y)
{
  extern __shared__ char smem[];
  u16* Cs  = (u16*)smem;               // [128][72]
  u16* Bs  = Cs + 128*72;              // [128][72]   (row m, n contiguous)
  u16* xTs = Bs + 128*72;              // [64][130]   x^T: [p][m]
  u16* Ws  = xTs + 64*130;             // [128][136]  weighted scores
  float* cum_s = (float*)(Ws + 128*136);
  int h = blockIdx.x, c = blockIdx.y, b = blockIdx.z;
  int tid = threadIdx.x, wv=tid>>5, lane=tid&31, hl=lane>>4, lr=lane&15;
  int t0 = b*TT + c*CHUNK_L;

  for (int i=tid;i<128*64;i+=256){ int l=i>>6, n=i&63; Cs[l*72+n]=Crb[(size_t)(t0+l)*64+n]; }
  if (tid<128) cum_s[tid] = cum[(size_t)(t0+tid)*32 + h];
  v8f yacc[4]; for (int j=0;j<4;j++) yacc[j]=zero8();
  __syncthreads();

  for (int vr=0; vr<4; ++vr){
    int vb = vr>>1, r = vr&1;
    const u16* xv = vb ? xb : xg;
    for (int i=tid;i<128*64;i+=256){
      int m=i>>6, n=i&63;
      int tm = t0 + m;
      u16 val;
      if (vb){ int tbm = c*CHUNK_L + m;
               val = (tbm==0) ? (u16)0 : Brb[(size_t)(tm-1)*128 + r*64 + n]; }
      else     val = Brb[(size_t)tm*128 + r*64 + n];
      Bs[m*72+n] = val;
    }
    for (int i=tid;i<64*128;i+=256){
      int p=i&63, m=i>>6;
      xTs[p*130+m] = xv[(size_t)(t0+m)*4096 + h*128 + r*64 + p];
    }
    __syncthreads();
    // phase 1: scores (wave owns 16-row strip, all 8 col tiles), K = n (64)
    v8f sacc[8]; for (int j=0;j<8;j++) sacc[j]=zero8();
    for (int kk=0;kk<64;kk+=32){
      v16bf af = ldsFragA(Cs, 72, wv*16+lr, kk, hl);
      for (int j=0;j<8;j++){
        v16bf bfv = ldsFragB(Bs, 72, j*16+lr, kk, hl);
        sacc[j] = wmma_bf16(af, bfv, sacc[j]);
      }
    }
    // decay mask + SCALE in C/D register layout; restripe via LDS to A layout
    for (int j=0;j<8;j++){
#pragma unroll
      for (int jj=0;jj<8;jj++){
        int l = wv*16 + jj + hl*8;
        int m = j*16 + lr;
        float w = (m<=l) ? __expf(cum_s[l]-cum_s[m])*SCALE_F : 0.f;
        Ws[l*136+m] = f2bf(sacc[j][jj]*w);
      }
    }
    __syncthreads();
    // phase 2: Y += W x xv, K = m (128), N = p (64 -> 4 tiles)
    for (int kk=0;kk<128;kk+=32){
      v16bf af = ldsFragA(Ws, 136, wv*16+lr, kk, hl);
      for (int j=0;j<4;j++){
        v16bf bfv = ldsFragB(xTs, 130, j*16+lr, kk, hl);
        yacc[j] = wmma_bf16(af, bfv, yacc[j]);
      }
    }
    __syncthreads();
  }
  for (int j=0;j<4;j++){
#pragma unroll
    for (int jj=0;jj<8;jj++){
      int l = wv*16 + jj + hl*8;
      int p = j*16 + lr;
      y[(size_t)(t0+l)*D_INNER + h*64 + p] = yacc[j][jj];
    }
  }
}

// ---------------- chunk end-state: st[n,p] = sum_{v,r,l} dec*B*x -----------
__global__ __launch_bounds__(256) void state_kernel(
    const u16* __restrict__ Brb, const u16* __restrict__ xg, const u16* __restrict__ xb,
    const float* __restrict__ cum, float* __restrict__ st, float* __restrict__ cdec)
{
  extern __shared__ char smem[];
  u16* As  = (u16*)smem;        // [64][130]  A[n][l] = B[l,r,n]*dec_end[l]
  u16* xTs = As + 64*130;       // [64][130]  x^T [p][l]
  float* cum_s = (float*)(xTs + 64*130);
  int h=blockIdx.x, c=blockIdx.y, b=blockIdx.z;
  int tid=threadIdx.x, wv=tid>>5, lane=tid&31, hl=lane>>4, lr=lane&15;
  int t0 = b*TT + c*CHUNK_L;
  if (tid<128) cum_s[tid]=cum[(size_t)(t0+tid)*32+h];
  __syncthreads();
  float cum_last = cum_s[127];
  int ti = wv>>1, tjb = (wv&1)*2;     // 8 waves -> 2 output tiles each
  v8f acc[2]; acc[0]=zero8(); acc[1]=zero8();

  for (int vr=0; vr<4; ++vr){
    int vb=vr>>1, r=vr&1;
    const u16* xv = vb ? xb : xg;
    for (int i=tid;i<64*128;i+=256){
      int n=i&63, l=i>>6;
      int tm=t0+l;
      float bv;
      if (vb){ int tbm=c*CHUNK_L+l;
               bv = (tbm==0)?0.f:bf2f(Brb[(size_t)(tm-1)*128+r*64+n]); }
      else     bv = bf2f(Brb[(size_t)tm*128+r*64+n]);
      As[n*130+l] = f2bf(bv*__expf(cum_last - cum_s[l]));
    }
    for (int i=tid;i<64*128;i+=256){
      int p=i&63, l=i>>6;
      xTs[p*130+l] = xv[(size_t)(t0+l)*4096 + h*128 + r*64 + p];
    }
    __syncthreads();
    for (int kk=0;kk<128;kk+=32){
      v16bf af = ldsFragA(As, 130, ti*16+lr, kk, hl);
      for (int jt=0;jt<2;jt++){
        v16bf bfv = ldsFragB(xTs, 130, (tjb+jt)*16+lr, kk, hl);
        acc[jt] = wmma_bf16(af, bfv, acc[jt]);
      }
    }
    __syncthreads();
  }
  size_t base = (((size_t)(b*NCH+c))*N_HEADS + h)*4096;
  for (int jt=0;jt<2;jt++){
#pragma unroll
    for (int jj=0;jj<8;jj++){
      int n = ti*16 + jj + hl*8;
      int p = (tjb+jt)*16 + lr;
      st[base + n*64 + p] = acc[jt][jj];
    }
  }
  if (tid==0) cdec[(b*NCH+c)*N_HEADS + h] = __expf(cum_last);
}

// ---------------- sequential chunk scan ------------------------------------
__global__ __launch_bounds__(256) void scan_kernel(const float* __restrict__ st,
    const float* __restrict__ cdec, float* __restrict__ Sprev)
{
  int h=blockIdx.x, b=blockIdx.y, tid=threadIdx.x;
  float s[16];
#pragma unroll
  for (int k=0;k<16;k++) s[k]=0.f;
  for (int c=0;c<NCH;c++){
    size_t base = (((size_t)(b*NCH+c))*N_HEADS + h)*4096;
    float cd = cdec[(b*NCH+c)*N_HEADS+h];
#pragma unroll
    for (int k=0;k<16;k++){
      int e = tid + k*256;
      Sprev[base+e] = s[k];
      s[k] = cd*s[k] + st[base+e];
    }
  }
}

// ---------------- inter-chunk: y += SCALE * dec_in * (C x Sprev) -----------
__global__ __launch_bounds__(256) void inter_kernel(const u16* __restrict__ Crb,
    const float* __restrict__ Sprev, const float* __restrict__ cum, float* __restrict__ y)
{
  extern __shared__ char smem[];
  u16* Cs  = (u16*)smem;          // [128][72]
  u16* STs = Cs + 128*72;         // [64][72]  S^T [p][n] in bf16
  float* cum_s = (float*)(STs + 64*72);
  int h=blockIdx.x, c=blockIdx.y, b=blockIdx.z;
  int tid=threadIdx.x, wv=tid>>5, lane=tid&31, hl=lane>>4, lr=lane&15;
  int t0 = b*TT + c*CHUNK_L;
  for (int i=tid;i<128*64;i+=256){ int l=i>>6,n=i&63; Cs[l*72+n]=Crb[(size_t)(t0+l)*64+n]; }
  size_t sbase = (((size_t)(b*NCH+c))*N_HEADS+h)*4096;
  for (int i=tid;i<64*64;i+=256){ int p=i&63, n=i>>6; STs[p*72+n]=f2bf(Sprev[sbase + n*64 + p]); }
  if (tid<128) cum_s[tid]=cum[(size_t)(t0+tid)*32+h];
  __syncthreads();
  v8f acc[4]; for (int j=0;j<4;j++) acc[j]=zero8();
  for (int kk=0;kk<64;kk+=32){
    v16bf af = ldsFragA(Cs, 72, wv*16+lr, kk, hl);
    for (int j=0;j<4;j++){
      v16bf bfv = ldsFragB(STs, 72, j*16+lr, kk, hl);
      acc[j] = wmma_bf16(af, bfv, acc[j]);
    }
  }
  for (int j=0;j<4;j++){
#pragma unroll
    for (int jj=0;jj<8;jj++){
      int l=wv*16+jj+hl*8, p=j*16+lr;
      size_t yi = (size_t)(t0+l)*D_INNER + h*64 + p;
      y[yi] += SCALE_F*__expf(cum_s[l])*acc[j][jj];
    }
  }
}

// ---------------- final gate + RMS norm -> bf16 ----------------------------
__global__ __launch_bounds__(256) void gate_kernel(const float* __restrict__ proj,
    const float* __restrict__ y, u16* __restrict__ ybf)
{
  int bt = blockIdx.x, tid = threadIdx.x;
  __shared__ float red[256];
  float g[8]; float ss=0.f;
  for (int j=0;j<8;j++){
    int i = tid + j*256;
    float z  = proj[(size_t)bt*D_PROJ + i];
    float gv = y[(size_t)bt*D_INNER + i] * siluf_(z);
    g[j]=gv; ss += gv*gv;
  }
  red[tid]=ss; __syncthreads();
  for (int s=128; s>0; s>>=1){ if (tid<s) red[tid]+=red[tid+s]; __syncthreads(); }
  float sc = rsqrtf(red[0]*(1.f/2048.f) + EPS_F);
  for (int j=0;j<8;j++){
    int i = tid + j*256;
    ybf[(size_t)bt*D_INNER + i] = f2bf(g[j]*sc);
  }
}

// ---------------------------------------------------------------------------
extern "C" void kernel_launch(void* const* d_in, const int* in_sizes, int n_in,
                              void* d_out, int out_size, void* d_ws, size_t ws_size,
                              hipStream_t stream)
{
  (void)in_sizes; (void)n_in; (void)out_size; (void)ws_size;
  const float* x     = (const float*)d_in[0];
  const float* W_in  = (const float*)d_in[1];
  const float* A_log = (const float*)d_in[2];
  const float* dt_b  = (const float*)d_in[3];
  const float* B_nb  = (const float*)d_in[4];
  const float* C_nb  = (const float*)d_in[5];
  const float* W_out = (const float*)d_in[6];
  float* out = (float*)d_out;

  char* base = (char*)d_ws; size_t off = 0;
  auto take = [&](size_t bytes)->void*{
    void* p = base + off;
    off = (off + bytes + 255) & ~(size_t)255;
    return p;
  };
  float* proj   = (float*)take((size_t)BT*D_PROJ*4);
  u16*   xbf    = (u16*)  take((size_t)BT*D_MODEL*2);
  u16*   Winb   = (u16*)  take((size_t)D_MODEL*D_PROJ*2);
  u16*   Woutb  = (u16*)  take((size_t)D_INNER*D_MODEL*2);
  float* dt_ws  = (float*)take((size_t)BT*32*4);
  float* lam_ws = (float*)take((size_t)BT*32*4);
  float* ldA_ws = (float*)take((size_t)BT*32*4);
  float* dtmean = (float*)take((size_t)BT*4);
  float* Bn_ws  = (float*)take((size_t)BT*128*4);
  float* Cn_ws  = (float*)take((size_t)BT*64*4);
  float* cosw   = (float*)take((size_t)BT*32*4);
  float* sinw   = (float*)take((size_t)BT*32*4);
  u16*   Crb    = (u16*)  take((size_t)BT*64*2);
  u16*   Brb    = (u16*)  take((size_t)BT*128*2);
  u16*   xgb    = (u16*)  take((size_t)BT*4096*2);
  u16*   xbb    = (u16*)  take((size_t)BT*4096*2);
  float* cum_ws = (float*)take((size_t)BT*32*4);
  float* y_ws   = (float*)take((size_t)BT*D_INNER*4);
  float* st_ws  = (float*)take((size_t)BB*NCH*N_HEADS*4096*4);
  float* Spv    = (float*)take((size_t)BB*NCH*N_HEADS*4096*4);
  float* cdec   = (float*)take((size_t)BB*NCH*N_HEADS*4);
  u16*   ybfb   = (u16*)  take((size_t)BT*D_INNER*2);

  // 1) bf16 conversions for WMMA GEMM inputs
  cvt_bf16_kernel<<<(BT*D_MODEL+255)/256, 256, 0, stream>>>(x, xbf, BT*D_MODEL);
  cvt_bf16_kernel<<<(D_MODEL*D_PROJ+255)/256, 256, 0, stream>>>(W_in, Winb, D_MODEL*D_PROJ);
  cvt_bf16_kernel<<<(D_INNER*D_MODEL+255)/256, 256, 0, stream>>>(W_out, Woutb, D_INNER*D_MODEL);

  // 2) proj = x @ W_in  (4096 x 6432 x 1024), 256x128 block tiles
  gemm_bf16_kernel<<<dim3((D_PROJ+127)/128, BT/256), 256, 0, stream>>>(
      xbf, Winb, proj, BT, D_PROJ, D_MODEL);

  // 3) elementwise prep, angle cumsum, RoPE + gates, chunk cumsum
  prep1_kernel<<<BT, 64, 0, stream>>>(proj, A_log, dt_b, B_nb, C_nb,
      dt_ws, lam_ws, ldA_ws, dtmean, Bn_ws, Cn_ws);
  ang_kernel<<<BB, 32, 0, stream>>>(proj, dtmean, cosw, sinw);
  prep2_kernel<<<BT, 128, 0, stream>>>(proj, Bn_ws, Cn_ws, cosw, sinw,
      dt_ws, lam_ws, ldA_ws, Crb, Brb, xgb, xbb);
  chunk_cum_kernel<<<BT/CHUNK_L, 32, 0, stream>>>(ldA_ws, cum_ws);

  // 4) intra-chunk attention (WMMA), per (head, chunk, batch)
  intra_kernel<<<dim3(N_HEADS, NCH, BB), 256, 88832, stream>>>(
      Crb, Brb, xgb, xbb, cum_ws, y_ws);

  // 5) chunk states (WMMA) + sequential scan + inter-chunk (WMMA)
  state_kernel<<<dim3(N_HEADS, NCH, BB), 256, 33792, stream>>>(
      Brb, xgb, xbb, cum_ws, st_ws, cdec);
  scan_kernel<<<dim3(N_HEADS, BB), 256, 0, stream>>>(st_ws, cdec, Spv);
  inter_kernel<<<dim3(N_HEADS, NCH, BB), 256, 28160, stream>>>(
      Crb, Spv, cum_ws, y_ws);

  // 6) gated RMS norm -> bf16, then out = y @ W_out (4096 x 1024 x 2048)
  gate_kernel<<<BT, 256, 0, stream>>>(proj, y_ws, ybfb);
  gemm_bf16_kernel<<<dim3((D_MODEL+127)/128, BT/256), 256, 0, stream>>>(
      ybfb, Woutb, out, BT, D_MODEL, D_INNER);
}
